// DeformableAttention_52286931862250
// MI455X (gfx1250) — matrix operand order repaired and checked by reading, not compiled, not run on hardware
//
#include <hip/hip_runtime.h>
#include <math.h>

// ---------------------------------------------------------------------------
// DeformableAttention on MI455X (gfx1250, wave32, WMMA).
// Sizes fixed by the reference: B=4, C=256, H=W=64, L=4096, D=64 (Q dim).
// d_out = [ out (B,L,C) fp32 | Q (B,L,64) fp32 ]
// d_ws  = [ feat (B*L f32) | Qb bf16 [b][l][64] | Vt bf16 [b][c][l] ]
// ---------------------------------------------------------------------------

#define BN   4
#define CN   256
#define HN   64
#define WN   64
#define LN   4096
#define DN   64

typedef __attribute__((ext_vector_type(16))) __bf16        v16bf;
typedef __attribute__((ext_vector_type(8)))  float         f32x8;
typedef __attribute__((ext_vector_type(4)))  unsigned int  u32x4;

union BPack { u32x4 u[2]; v16bf v; };   // 32 bytes: one WMMA bf16 operand slice

static __device__ __forceinline__ unsigned short f2bf(float f) {
    unsigned u = __float_as_uint(f);
    unsigned r = (u + 0x7FFFu + ((u >> 16) & 1u)) >> 16;   // RNE
    return (unsigned short)r;
}

// ---------------- kernel 1: feat[b,l] = mean_c x[b,c,l] ----------------
__global__ void k_mean(const float* __restrict__ x, float* __restrict__ feat) {
    int t = blockIdx.x * 256 + threadIdx.x;       // b*L + l
    int b = t >> 12, l = t & (LN - 1);
    const float* p = x + (size_t)b * CN * LN + l;
    float s = 0.f;
#pragma unroll 8
    for (int c = 0; c < CN; ++c) s += p[(size_t)c * LN];
    feat[t] = s * (1.f / 256.f);
}

// ------------- kernel 2: conv1d(k=3,pad=1) over flat L -> Q ------------
__global__ void k_q(const float* __restrict__ feat,
                    const float* __restrict__ w,     // (64,1,3)
                    const float* __restrict__ bias,  // (64,)
                    float* __restrict__ Qf,          // (B,L,64) fp32 (d_out Q region)
                    unsigned short* __restrict__ Qb) // (B,L,64) bf16
{
    int t = blockIdx.x * 256 + threadIdx.x;       // b*L + l
    int l = t & (LN - 1);
    float fm = (l > 0)      ? feat[t - 1] : 0.f;
    float f0 = feat[t];
    float fp = (l < LN - 1) ? feat[t + 1] : 0.f;
#pragma unroll 4
    for (int o = 0; o < DN; ++o) {
        float q = fmaf(w[3 * o], fm, fmaf(w[3 * o + 1], f0, fmaf(w[3 * o + 2], fp, bias[o])));
        Qf[(size_t)t * DN + o] = q;
        Qb[(size_t)t * DN + o] = f2bf(q);
    }
}

// ------ kernel 3: bilinear grid-sample -> Vt[b][c][l] (bf16, ch-major) ------
__global__ void k_sample(const float* __restrict__ x,
                         const float* __restrict__ Qf,
                         unsigned short* __restrict__ Vt) {
    int l = blockIdx.x * 256 + threadIdx.x;
    int c = blockIdx.y, b = blockIdx.z;
    float gx = Qf[((size_t)(b * LN + l)) * DN + 1];
    float gy = Qf[((size_t)(b * LN + l)) * DN + 0];
    float px = (gx + 1.f) * 0.5f * (WN - 1);
    float py = (gy + 1.f) * 0.5f * (HN - 1);
    float x0 = floorf(px), y0 = floorf(py);
    float wx1 = px - x0, wx0 = 1.f - wx1;
    float wy1 = py - y0, wy0 = 1.f - wy1;
    const float* xc = x + ((size_t)(b * CN + c)) * LN;

    auto tap = [&](float yi, float xi) -> float {
        bool v = (xi >= 0.f) & (xi < (float)WN) & (yi >= 0.f) & (yi < (float)HN);
        int xii = (int)fminf(fmaxf(xi, 0.f), (float)(WN - 1));
        int yii = (int)fminf(fmaxf(yi, 0.f), (float)(HN - 1));
        return v ? xc[yii * WN + xii] : 0.f;
    };
    float r = tap(y0, x0) * (wy0 * wx0) + tap(y0, x0 + 1.f) * (wy0 * wx1)
            + tap(y0 + 1.f, x0) * (wy1 * wx0) + tap(y0 + 1.f, x0 + 1.f) * (wy1 * wx1);
    Vt[((size_t)(b * CN + c)) * LN + l] = f2bf(r);
}

// --------- kernel 4: flash attention, WMMA bf16, fp32 accumulate ---------
// block = (16-query tile, batch); 2 waves, wave w owns channels [128w, 128w+128)
// key block = 64 keys/iteration: 8 S-WMMAs + 16 PV-WMMAs per wave per iter.
__global__ __launch_bounds__(64, 2) void k_attn(
    const unsigned short* __restrict__ Qb,  // [b][l][64] bf16 (queries & keys)
    const unsigned short* __restrict__ Vt,  // [b][c][l]  bf16 (values, ch-major)
    float* __restrict__ out)                // [b][l][256] fp32
{
    const int qt = blockIdx.x;              // 0..255
    const int b  = blockIdx.y;              // 0..3
    const int wv = threadIdx.x >> 5;        // wave in block -> channel slice
    const int lane = threadIdx.x & 31;
    const int n  = lane & 15;
    const int hi = lane >> 4;
    const int q0 = qt * 16;
    const int c0 = wv * 128;

    __shared__ __align__(16) unsigned short pstage[2][16][64];  // P transpose stage

    // --- A-operand: Q tile 16x64, two K-steps of 32 (contiguous K: k = 16*hi + s) ---
    const unsigned short* qrow = Qb + ((size_t)(b * LN + q0 + n)) * DN;
    BPack aq[2];
#pragma unroll
    for (int s = 0; s < 2; ++s) {
        aq[s].u[0] = *(const u32x4*)(qrow + s * 32 + hi * 16);
        aq[s].u[1] = *(const u32x4*)(qrow + s * 32 + hi * 16 + 8);
    }

    f32x8 acc[8];
#pragma unroll
    for (int nt = 0; nt < 8; ++nt) acc[nt] = (f32x8){0.f,0.f,0.f,0.f,0.f,0.f,0.f,0.f};
    float mrun[8], lrun[8];
#pragma unroll
    for (int r = 0; r < 8; ++r) { mrun[r] = -INFINITY; lrun[r] = 0.f; }

    for (int kb = 0; kb < LN; kb += 64) {
        // prefetch next key block (keys + values) into cache
        if (kb + 64 < LN) {
            __builtin_prefetch(Qb + ((size_t)(b * LN + kb + 64 + n)) * DN, 0, 1);
            __builtin_prefetch(Vt + ((size_t)(b * CN + c0 + n)) * LN + kb + 64, 0, 1);
        }

        // --- S = Q * K^T : four 16x16 tiles (64 keys), K-dim 64 = 2 wmma steps ---
        f32x8 s[4];
#pragma unroll
        for (int t = 0; t < 4; ++t) {
            const unsigned short* krow = Qb + ((size_t)(b * LN + kb + t * 16 + n)) * DN;
            BPack bk0, bk1;                       // B 32x16: lane=n, K split at 16 by hi
            bk0.u[0] = *(const u32x4*)(krow + hi * 16);
            bk0.u[1] = *(const u32x4*)(krow + hi * 16 + 8);
            bk1.u[0] = *(const u32x4*)(krow + 32 + hi * 16);
            bk1.u[1] = *(const u32x4*)(krow + 32 + hi * 16 + 8);
            f32x8 z = (f32x8){0.f,0.f,0.f,0.f,0.f,0.f,0.f,0.f};
            s[t] = __builtin_amdgcn_wmma_f32_16x16x32_bf16(false, aq[0].v, false, bk0.v,
                                                           (short)0, z,    false, false);
            s[t] = __builtin_amdgcn_wmma_f32_16x16x32_bf16(false, aq[1].v, false, bk1.v,
                                                           (short)0, s[t], false, false);
        }

        // --- online softmax; rows live in VGPR index r (row = r + 8*hi) ---
        float p[4][8];
#pragma unroll
        for (int r = 0; r < 8; ++r) {
            float m = fmaxf(fmaxf(s[0][r], s[1][r]), fmaxf(s[2][r], s[3][r]));
            m = fmaxf(m, __shfl_xor(m, 1, 32));
            m = fmaxf(m, __shfl_xor(m, 2, 32));
            m = fmaxf(m, __shfl_xor(m, 4, 32));
            m = fmaxf(m, __shfl_xor(m, 8, 32));
            float nm = fmaxf(mrun[r], m);
            float al = __expf(mrun[r] - nm);
#pragma unroll
            for (int t = 0; t < 4; ++t) p[t][r] = __expf(s[t][r] - nm);
            float rs = (p[0][r] + p[1][r]) + (p[2][r] + p[3][r]);
            rs += __shfl_xor(rs, 1, 32);
            rs += __shfl_xor(rs, 2, 32);
            rs += __shfl_xor(rs, 4, 32);
            rs += __shfl_xor(rs, 8, 32);
            lrun[r] = lrun[r] * al + rs;
            mrun[r] = nm;
#pragma unroll
            for (int nt = 0; nt < 8; ++nt) acc[nt][r] *= al;
        }

        // --- P (C/D layout) -> LDS -> A layout (16x64 bf16, two K-steps) ---
#pragma unroll
        for (int r = 0; r < 8; ++r)
#pragma unroll
            for (int t = 0; t < 4; ++t)
                pstage[wv][r + 8 * hi][t * 16 + n] = f2bf(p[t][r]);
        asm volatile("s_wait_dscnt 0" ::: "memory");   // same-wave LDS RAW
        BPack pa[2];
        const unsigned short* prow = &pstage[wv][n][0];
#pragma unroll
        for (int s2 = 0; s2 < 2; ++s2) {
            pa[s2].u[0] = *(const u32x4*)(prow + s2 * 32 + hi * 16);
            pa[s2].u[1] = *(const u32x4*)(prow + s2 * 32 + hi * 16 + 8);
        }

        // --- O += P * V : 8 channel tiles of 16, K=64 keys = 2 wmma steps ---
#pragma unroll
        for (int nt = 0; nt < 8; ++nt) {
            const unsigned short* vrow =
                Vt + ((size_t)(b * CN + c0 + nt * 16 + n)) * LN + kb;
            BPack bv0, bv1;
            bv0.u[0] = *(const u32x4*)(vrow + hi * 16);
            bv0.u[1] = *(const u32x4*)(vrow + hi * 16 + 8);
            bv1.u[0] = *(const u32x4*)(vrow + 32 + hi * 16);
            bv1.u[1] = *(const u32x4*)(vrow + 32 + hi * 16 + 8);
            acc[nt] = __builtin_amdgcn_wmma_f32_16x16x32_bf16(false, pa[0].v, false, bv0.v,
                                                              (short)0, acc[nt], false, false);
            acc[nt] = __builtin_amdgcn_wmma_f32_16x16x32_bf16(false, pa[1].v, false, bv1.v,
                                                              (short)0, acc[nt], false, false);
        }
    }

    // --- epilogue: out = acc / l ---
#pragma unroll
    for (int r = 0; r < 8; ++r) {
        float inv = 1.0f / lrun[r];
        float* orow = out + ((size_t)(b * LN + q0 + r + 8 * hi)) * CN + c0 + n;
#pragma unroll
        for (int nt = 0; nt < 8; ++nt) orow[nt * 16] = acc[nt][r] * inv;
    }
}

// ---------------------------------------------------------------------------
extern "C" void kernel_launch(void* const* d_in, const int* in_sizes, int n_in,
                              void* d_out, int out_size, void* d_ws, size_t ws_size,
                              hipStream_t stream) {
    (void)in_sizes; (void)n_in; (void)out_size; (void)ws_size;
    const float* x     = (const float*)d_in[0];   // (B,C,H,W)
    const float* w_off = (const float*)d_in[1];   // (64,1,3)
    const float* b_off = (const float*)d_in[2];   // (64,)

    float* out = (float*)d_out;                          // (B,L,C)
    float* Qf  = out + (size_t)BN * LN * CN;             // (B,L,64)

    char* ws = (char*)d_ws;
    float*          feat = (float*)ws;                                   // 64 KB
    unsigned short* Qb   = (unsigned short*)(ws + 65536);                // 2 MB
    unsigned short* Vt   = (unsigned short*)(ws + 65536 + 2097152);      // 8 MB

    k_mean  <<<dim3((BN * LN) / 256), 256, 0, stream>>>(x, feat);
    k_q     <<<dim3((BN * LN) / 256), 256, 0, stream>>>(feat, w_off, b_off, Qf, Qb);
    k_sample<<<dim3(LN / 256, CN, BN), 256, 0, stream>>>(x, Qf, Vt);
    k_attn  <<<dim3(LN / 16, BN), 64, 0, stream>>>(Qb, Vt, out);
}
